// TransformerBlockTreeGNN_77970836291714
// MI455X (gfx1250) — matrix operand order repaired and checked by reading, not compiled, not run on hardware
//
#include <hip/hip_runtime.h>
#include <hip/hip_bf16.h>
#include <stdint.h>

// ---------------- problem constants ----------------
constexpr int Bq  = 4;
constexpr int Lq  = 1024;
constexpr int Dq  = 512;
constexpr int Hq  = 8;
constexpr int BLq = Bq * Lq;         // 4096 rows
constexpr int Kgemm = 512;           // contraction dim for every GEMM in this model
constexpr float NEGF = -3.4028235e38f;

// ---------------- types ----------------
typedef __bf16 bf16;
typedef __attribute__((ext_vector_type(8)))  __bf16 bh8;
typedef __attribute__((ext_vector_type(16))) __bf16 bh16;
typedef __attribute__((ext_vector_type(8)))  float  f32x8;

static __device__ __forceinline__ bh16 mk16(bh8 lo, bh8 hi) {
  bh16 r;
#pragma unroll
  for (int i = 0; i < 8; ++i) { r[i] = lo[i]; r[8 + i] = hi[i]; }
  return r;
}

static __device__ __forceinline__ f32x8 wmma_bf16(bh16 a, bh16 b, f32x8 c) {
  // D = A(16x32 bf16) * B(32x16 bf16) + C(16x16 f32)
  return __builtin_amdgcn_wmma_f32_16x16x32_bf16(false, a, false, b, (short)0, c,
                                                 false, false);
}

static __device__ __forceinline__ bh16 ld_a16(const bf16* p) {   // A frag: +0,+16
  return mk16(*(const bh8*)(p), *(const bh8*)(p + 16));
}
static __device__ __forceinline__ bh16 ld_b16(const bf16* p) {   // B frag: +0,+16
  return mk16(*(const bh8*)(p), *(const bh8*)(p + 16));
}

// ---------------- weight cast + transpose: W[KxN] f32 -> Wt[NxK] bf16 ----------------
__global__ __launch_bounds__(256) void castT_kernel(const float* __restrict__ W,
                                                    bf16* __restrict__ Wt,
                                                    int K, int N) {
  int idx = blockIdx.x * 256 + threadIdx.x;
  if (idx >= K * N) return;
  int k = idx / N, n = idx - k * N;
  Wt[(size_t)n * K + k] = (bf16)W[idx];
}

// ---------------- ancestor chains (8 levels) ----------------
__global__ __launch_bounds__(256) void anc_kernel(const int* __restrict__ parents,
                                                  int* __restrict__ anc) {
  int idx = blockIdx.x * 256 + threadIdx.x;  // b*L + l
  if (idx >= BLq) return;
  int b = idx >> 10, l = idx & (Lq - 1);
  int cur = l;
#pragma unroll
  for (int k = 0; k < 8; ++k) {
    anc[idx * 8 + k] = cur;
    int p = (cur >= 0) ? parents[b * Lq + cur] : -1;
    cur = (p >= 0 && p < Lq) ? p : -1;
  }
}

// ---------------- bucketized tree distance (min(d,7)) via LCA over 8-level chains -----
__global__ __launch_bounds__(256) void bucket_kernel(const int* __restrict__ anc,
                                                     uint8_t* __restrict__ buckets) {
  int bi = blockIdx.x;              // b*L + i
  int b = bi >> 10;
  int ai[8];
#pragma unroll
  for (int k = 0; k < 8; ++k) ai[k] = anc[(size_t)bi * 8 + k];
  size_t rowbase = (size_t)bi * Lq;
  for (int j = threadIdx.x; j < Lq; j += 256) {
    int aj[8];
#pragma unroll
    for (int k = 0; k < 8; ++k) aj[k] = anc[((size_t)b * Lq + j) * 8 + k];
    int dmin = 255;
#pragma unroll
    for (int a = 0; a < 8; ++a) {
      int va = ai[a];
#pragma unroll
      for (int c = 0; c < 8; ++c) {
        if (va >= 0 && va == aj[c]) dmin = min(dmin, a + c);
      }
    }
    buckets[rowbase + j] = (uint8_t)min(dmin, 7);
  }
}

// ---------------- LayerNorm -> bf16 (one block per row of 512) ----------------
__global__ __launch_bounds__(256) void ln_cast_kernel(const float* __restrict__ x,
                                                      const float* __restrict__ g,
                                                      const float* __restrict__ bb,
                                                      bf16* __restrict__ out) {
  int row = blockIdx.x, t = threadIdx.x;
  const float* xr = x + (size_t)row * Dq;
  float e0 = xr[t], e1 = xr[t + 256];
  __shared__ float rs[256], rq[256];
  rs[t] = e0 + e1;
  rq[t] = e0 * e0 + e1 * e1;
  __syncthreads();
  for (int o = 128; o > 0; o >>= 1) {
    if (t < o) { rs[t] += rs[t + o]; rq[t] += rq[t + o]; }
    __syncthreads();
  }
  float mean = rs[0] * (1.f / Dq);
  float var  = rq[0] * (1.f / Dq) - mean * mean;
  float inv  = rsqrtf(var + 1e-5f);
  out[(size_t)row * Dq + t]       = (bf16)((e0 - mean) * inv * g[t] + bb[t]);
  out[(size_t)row * Dq + t + 256] = (bf16)((e1 - mean) * inv * g[t + 256] + bb[t + 256]);
}

// ---------------- bf16 WMMA GEMM: C[MxN] = A[MxK] * Wt[NxK]^T + bias ----------------
// Each wave computes a 32x64 output block: 2 M-subtiles x 4 N-subtiles = 8 WMMA
// accumulators; B fragments reused across both M-subtiles. Register double-buffered
// k-loop (static K=512) so next tile's 12 b128 loads overlap current 8 WMMAs.
__global__ __launch_bounds__(256) void gemm32_kernel(
    const bf16* __restrict__ A, int lda,
    const bf16* __restrict__ Bt, int ldb,          // [N x K] row-major (pre-transposed W)
    const float* __restrict__ bias,                // [N]
    const float* __restrict__ residual,            // nullable, [M x ldF]
    float* __restrict__ outF,                      // nullable, [M x ldF]
    bf16* __restrict__ outH,                       // nullable, [M x ldH] at colOff
    int M, int N, int ldH, int colOff, int ldF) {
  const int wave = threadIdx.x >> 5;
  const int lane = threadIdx.x & 31;
  const int gw = blockIdx.x * 8 + wave;
  const int tilesN = N >> 6;
  const int totalW = (M >> 5) * tilesN;
  if (gw >= totalW) return;
  const int tm = gw / tilesN, tn = gw - tm * tilesN;
  const int half = lane >> 4, lidx = lane & 15;

  const bf16* Arow0 = A + (size_t)(tm * 32 + lidx) * lda + half * 8;
  const bf16* Arow1 = Arow0 + (size_t)16 * lda;
  const bf16* Brow  = Bt + (size_t)(tn * 64 + lidx) * ldb + half * 8;
  const size_t bstep = (size_t)16 * ldb;

  f32x8 acc[2][4] = {};
  bh16 a[2], b[4], an[2], bn[4];
  a[0] = ld_a16(Arow0);
  a[1] = ld_a16(Arow1);
#pragma unroll
  for (int s = 0; s < 4; ++s) b[s] = ld_b16(Brow + s * bstep);

#pragma unroll 2
  for (int kc = 32; kc < Kgemm; kc += 32) {
    // issue next tile's loads before this tile's WMMAs (pipelined)
    an[0] = ld_a16(Arow0 + kc);
    an[1] = ld_a16(Arow1 + kc);
#pragma unroll
    for (int s = 0; s < 4; ++s) bn[s] = ld_b16(Brow + s * bstep + kc);
    __builtin_prefetch(Arow0 + kc + 32, 0, 3);
#pragma unroll
    for (int i = 0; i < 2; ++i)
#pragma unroll
      for (int s = 0; s < 4; ++s) acc[i][s] = wmma_bf16(a[i], b[s], acc[i][s]);
    a[0] = an[0];
    a[1] = an[1];
#pragma unroll
    for (int s = 0; s < 4; ++s) b[s] = bn[s];
  }
#pragma unroll
  for (int i = 0; i < 2; ++i)
#pragma unroll
    for (int s = 0; s < 4; ++s) acc[i][s] = wmma_bf16(a[i], b[s], acc[i][s]);

#pragma unroll
  for (int i = 0; i < 2; ++i)
#pragma unroll
    for (int s = 0; s < 4; ++s)
#pragma unroll
      for (int r = 0; r < 8; ++r) {
        int row = tm * 32 + i * 16 + r + half * 8;
        int col = tn * 64 + s * 16 + lidx;
        float v = acc[i][s][r] + bias[col];
        if (residual) v += residual[(size_t)row * ldF + col];
        if (outF) outF[(size_t)row * ldF + col] = v;
        if (outH) outH[(size_t)row * ldH + colOff + col] = (bf16)v;
      }
}

// ---------------- fused flash attention ----------------
// MODE 0: logits/8 + bias_table[h][bucket[b,i,j]], masked by pad (tree attention)
// MODE 1: logits/8 masked to adjacency (parent/child/self) & pad (GAT)
// qkv layout: [B*L x 1536] bf16, cols 0..511 = Q, 512..1023 = K, 1024..1535 = V
// One wave handles a 16-query tile of one (b,h) and streams all keys in tiles of 32.
struct AttnScr {
  bf16 vt[64 * 32];   // V tile transposed: vt[d*32 + j]
  bf16 p[16 * 32];    // probability tile, row-major
  float tbl[8];       // bias table row for this head
};

template <int MODE>
__global__ __launch_bounds__(256) void attn16_kernel(
    const bf16* __restrict__ qkv,
    const uint8_t* __restrict__ buckets,
    const float* __restrict__ btab,
    const int* __restrict__ parents,
    const uint8_t* __restrict__ pad,
    bf16* __restrict__ Y) {
  __shared__ AttnScr scr[8];
  const int wave = threadIdx.x >> 5;
  const int lane = threadIdx.x & 31;
  AttnScr& S = scr[wave];
  const int gw = blockIdx.x * 8 + wave;
  const int qt = gw & 63;            // L/16 = 64 query tiles
  const int h  = (gw >> 6) & 7;
  const int b  = gw >> 9;
  const int half = lane >> 4, lidx = lane & 15;
  const size_t baseBL = (size_t)b * Lq;
  const int i0 = qt * 16;

  if (MODE == 0 && lane < 8) S.tbl[lane] = btab[h * 8 + lane];

  // Q in WMMA-A layout (d=0..31 and d=32..63 chunks)
  const bf16* Qrow = qkv + (baseBL + i0 + lidx) * 1536 + h * 64 + half * 8;
  bh16 aq0 = mk16(*(const bh8*)(Qrow + 0),  *(const bh8*)(Qrow + 16));
  bh16 aq1 = mk16(*(const bh8*)(Qrow + 32), *(const bh8*)(Qrow + 48));

  float mrun[8], lrun[8];
  int par_q[8];
  bool pad_q[8];
  f32x8 ya[4] = {};
#pragma unroll
  for (int r = 0; r < 8; ++r) {
    int i = i0 + r + half * 8;
    pad_q[r] = pad[baseBL + i] != 0;
    par_q[r] = (MODE == 1) ? parents[baseBL + i] : 0;
    mrun[r] = NEGF;
    lrun[r] = 0.f;
  }

  for (int jb = 0; jb < Lq; jb += 32) {
    // ---- cluster all global loads for this tile up front ----
    // K fragments for both 16-key subtiles (8 b128 loads, one wait point)
    bh16 bk[2][2];
#pragma unroll
    for (int nt = 0; nt < 2; ++nt) {
      const bf16* Krow = qkv + (baseBL + jb + nt * 16 + lidx) * 1536 + 512 +
                         h * 64 + half * 8;
      bk[nt][0] = mk16(*(const bh8*)(Krow + 0),  *(const bh8*)(Krow + 16));
      bk[nt][1] = mk16(*(const bh8*)(Krow + 32), *(const bh8*)(Krow + 48));
    }
    // stage V tile (32 keys x 64 dims) transposed into this wave's LDS slice
    {
      const bf16* Vrow = qkv + (baseBL + jb + lane) * 1536 + 1024 + h * 64;
#pragma unroll
      for (int c = 0; c < 8; ++c) {
        bh8 v = *(const bh8*)(Vrow + c * 8);
#pragma unroll
        for (int t = 0; t < 8; ++t) S.vt[(c * 8 + t) * 32 + lane] = v[t];
      }
      if (jb + 32 < Lq)  // pull next tile's K/V rows toward the WGP
        __builtin_prefetch(Vrow + 32 * 1536 - 512, 0, 3);
    }
    // ---- logits S = Q * K^T for 16x32 tile (two 16x16 WMMA output tiles) ----
    f32x8 s01[2];
#pragma unroll
    for (int nt = 0; nt < 2; ++nt) {
      f32x8 s = {};
      s = wmma_bf16(aq0, bk[nt][0], s);
      s = wmma_bf16(aq1, bk[nt][1], s);
      s01[nt] = s;
    }
    // ---- bias / mask ----
#pragma unroll
    for (int nt = 0; nt < 2; ++nt) {
      int j = jb + nt * 16 + lidx;
      bool okj = pad[baseBL + j] != 0;
      int pj = (MODE == 1) ? parents[baseBL + j] : 0;
#pragma unroll
      for (int r = 0; r < 8; ++r) {
        int i = i0 + r + half * 8;
        float v = s01[nt][r] * 0.125f;  // 1/sqrt(DH)
        if (MODE == 0) {
          v += S.tbl[buckets[((size_t)b * Lq + i) * Lq + j]];
          if (!(pad_q[r] && okj)) v = NEGF;
        } else {
          bool adj = (i == j) | (par_q[r] == j) | (pj == i);
          if (!(adj && pad_q[r] && okj)) v = NEGF;
        }
        s01[nt][r] = v;
      }
    }
    // ---- online softmax (row reductions across 16 lanes; wave32) ----
    float scale[8];
#pragma unroll
    for (int r = 0; r < 8; ++r) {
      float tmax = fmaxf(s01[0][r], s01[1][r]);
#pragma unroll
      for (int m = 1; m < 16; m <<= 1) tmax = fmaxf(tmax, __shfl_xor(tmax, m, 32));
      float nm = fmaxf(mrun[r], tmax);
      scale[r] = __expf(mrun[r] - nm);
      mrun[r] = nm;
      float p0 = __expf(s01[0][r] - nm);
      float p1 = __expf(s01[1][r] - nm);
      int m_row = r + half * 8;
      S.p[m_row * 32 + lidx]      = (bf16)p0;
      S.p[m_row * 32 + 16 + lidx] = (bf16)p1;
      float ts = p0 + p1;
#pragma unroll
      for (int m = 1; m < 16; m <<= 1) ts += __shfl_xor(ts, m, 32);
      lrun[r] = lrun[r] * scale[r] + ts;
    }
#pragma unroll
    for (int nd = 0; nd < 4; ++nd)
#pragma unroll
      for (int r = 0; r < 8; ++r) ya[nd][r] *= scale[r];
    // ---- Y += P * V (P from LDS in A layout, V^T from LDS in B layout) ----
    const bf16* Pr = S.p + lidx * 32 + half * 8;
    bh16 ap = mk16(*(const bh8*)(Pr), *(const bh8*)(Pr + 16));
#pragma unroll
    for (int nd = 0; nd < 4; ++nd) {
      const bf16* Vr = S.vt + (nd * 16 + lidx) * 32 + half * 8;
      bh16 bv = mk16(*(const bh8*)(Vr), *(const bh8*)(Vr + 16));
      ya[nd] = wmma_bf16(ap, bv, ya[nd]);
    }
  }
  // ---- finalize: divide by row sums, store bf16 [B*L x 512] ----
#pragma unroll
  for (int nd = 0; nd < 4; ++nd)
#pragma unroll
    for (int r = 0; r < 8; ++r) {
      int i = i0 + r + half * 8;
      float val = ya[nd][r] / lrun[r];
      Y[(baseBL + i) * (size_t)Dq + h * 64 + nd * 16 + lidx] = (bf16)val;
    }
}

// ---------------- host orchestration ----------------
extern "C" void kernel_launch(void* const* d_in, const int* in_sizes, int n_in,
                              void* d_out, int out_size, void* d_ws, size_t ws_size,
                              hipStream_t stream) {
  (void)in_sizes; (void)n_in; (void)out_size; (void)ws_size;
  const float*   x       = (const float*)d_in[0];
  const int*     parents = (const int*)d_in[1];
  const uint8_t* pad     = (const uint8_t*)d_in[2];
  const float* ln1_g = (const float*)d_in[3];
  const float* ln1_b = (const float*)d_in[4];
  const float* qkv_w = (const float*)d_in[5];
  const float* qkv_b = (const float*)d_in[6];
  const float* ow    = (const float*)d_in[7];
  const float* ob    = (const float*)d_in[8];
  const float* btab  = (const float*)d_in[9];
  const float* gat_ln_g = (const float*)d_in[10];
  const float* gat_ln_b = (const float*)d_in[11];
  const float* g_wq = (const float*)d_in[12];
  const float* g_bq = (const float*)d_in[13];
  const float* g_wk = (const float*)d_in[14];
  const float* g_bk = (const float*)d_in[15];
  const float* g_wv = (const float*)d_in[16];
  const float* g_bv = (const float*)d_in[17];
  const float* g_wo = (const float*)d_in[18];
  const float* g_bo = (const float*)d_in[19];
  float* out = (float*)d_out;

  // ---- carve workspace ----
  char* ws = (char*)d_ws;
  size_t off = 0;
  auto take = [&](size_t n) -> char* {
    char* p = ws + off;
    off += (n + 255) & ~(size_t)255;
    return p;
  };
  int*     anc     = (int*)take((size_t)BLq * 8 * 4);
  uint8_t* buckets = (uint8_t*)take((size_t)Bq * Lq * Lq);
  bf16* wt_qkv = (bf16*)take((size_t)1536 * 512 * 2);
  bf16* wt_out = (bf16*)take((size_t)512 * 512 * 2);
  bf16* wt_g[2][4];
  for (int l = 0; l < 2; ++l)
    for (int m = 0; m < 4; ++m) wt_g[l][m] = (bf16*)take((size_t)512 * 512 * 2);
  bf16*  hbf  = (bf16*)take((size_t)BLq * 512 * 2);
  bf16*  qkvb = (bf16*)take((size_t)BLq * 1536 * 2);
  bf16*  ybf  = (bf16*)take((size_t)BLq * 512 * 2);
  float* xb0  = (float*)take((size_t)BLq * 512 * 4);
  float* xb1  = (float*)take((size_t)BLq * 512 * 4);

  // ---- one-time per call: cast/transpose weights, tree structure ----
  castT_kernel<<<(512 * 1536 + 255) / 256, 256, 0, stream>>>(qkv_w, wt_qkv, 512, 1536);
  castT_kernel<<<(512 * 512 + 255) / 256, 256, 0, stream>>>(ow, wt_out, 512, 512);
  for (int l = 0; l < 2; ++l) {
    const float* wsrc[4] = {g_wq + (size_t)l * 512 * 512, g_wk + (size_t)l * 512 * 512,
                            g_wv + (size_t)l * 512 * 512, g_wo + (size_t)l * 512 * 512};
    for (int m = 0; m < 4; ++m)
      castT_kernel<<<(512 * 512 + 255) / 256, 256, 0, stream>>>(wsrc[m], wt_g[l][m],
                                                                512, 512);
  }
  anc_kernel<<<BLq / 256, 256, 0, stream>>>(parents, anc);
  bucket_kernel<<<BLq, 256, 0, stream>>>(anc, buckets);

  // ---- stage 1: tree-biased attention ----
  ln_cast_kernel<<<BLq, 256, 0, stream>>>(x, ln1_g, ln1_b, hbf);
  // QKV: [4096x512] x [512x1536] -> bf16 qkv buffer
  gemm32_kernel<<<(BLq / 32) * (1536 / 64) / 8, 256, 0, stream>>>(
      hbf, 512, wt_qkv, 512, qkv_b, nullptr, nullptr, qkvb,
      BLq, 1536, 1536, 0, 512);
  attn16_kernel<0><<<(Bq * Hq * (Lq / 16)) / 8, 256, 0, stream>>>(
      qkvb, buckets, btab, parents, pad, ybf);
  // out-proj + residual -> xb0 (f32)
  gemm32_kernel<<<(BLq / 32) * (512 / 64) / 8, 256, 0, stream>>>(
      ybf, 512, wt_out, 512, ob, x, xb0, nullptr,
      BLq, 512, 0, 0, 512);

  // ---- GAT layers ----
  const float* xin = xb0;
  for (int l = 0; l < 2; ++l) {
    ln_cast_kernel<<<BLq, 256, 0, stream>>>(xin, gat_ln_g + (size_t)l * 512,
                                            gat_ln_b + (size_t)l * 512, hbf);
    const float* biases[3] = {g_bq + (size_t)l * 512, g_bk + (size_t)l * 512,
                              g_bv + (size_t)l * 512};
    for (int m = 0; m < 3; ++m) {
      gemm32_kernel<<<(BLq / 32) * (512 / 64) / 8, 256, 0, stream>>>(
          hbf, 512, wt_g[l][m], 512, biases[m], nullptr, nullptr, qkvb,
          BLq, 512, 1536, m * 512, 512);
    }
    attn16_kernel<1><<<(Bq * Hq * (Lq / 16)) / 8, 256, 0, stream>>>(
        qkvb, nullptr, nullptr, parents, pad, ybf);
    float* xoutp = (l == 0) ? xb1 : out;
    gemm32_kernel<<<(BLq / 32) * (512 / 64) / 8, 256, 0, stream>>>(
        ybf, 512, wt_g[l][3], 512, g_bo + (size_t)l * 512, xin, xoutp, nullptr,
        BLq, 512, 0, 0, 512);
    xin = xb1;
  }
}
